// LAGNet_28200755266180
// MI455X (gfx1250) — matrix operand order repaired and checked by reading, not compiled, not run on hardware
//
#include <hip/hip_runtime.h>

// ---------------------------------------------------------------------------
// LAGNet forward for MI455X (gfx1250, wave32).
// Activations bf16 channel-last; GEMMs via v_wmma_f32_16x16x32_bf16.
// Fragment layouts used here (contiguous k, not the ISA table order):
//   A (16x32, MxK): lane l -> row = l&15, h = l>>4; elems = K{16h .. 16h+15}
//   B (32x16, KxN): lane l -> col = l&15, h = l>>4; elems = K{16h .. 16h+15}
//   C/D (16x16)   : lane l -> col = l&15, h = l>>4; vgpr i -> row 8h+i
// ---------------------------------------------------------------------------

typedef __attribute__((ext_vector_type(8)))  __bf16 v8bf;
typedef __attribute__((ext_vector_type(16))) __bf16 v16bf;
typedef __attribute__((ext_vector_type(8)))  float  v8f;

#define B_   8
#define N_   2048
#define P_   (B_ * N_)       // 16384 points
#define K1_  16
#define K2_  8
#define L1_  (P_ * K1_)      // 262144
#define L2_  (P_ * K2_)      // 131072
#define PT_  8               // column tiles per wave
#define EPS_ 1e-5f

__device__ __forceinline__ v16bf cat16(v8bf lo, v8bf hi) {
  return __builtin_shufflevector(lo, hi, 0,1,2,3,4,5,6,7,8,9,10,11,12,13,14,15);
}

// One 16x16 f32 accumulator tile: Y[tm*16.., p..] += W(16xCx) * X(Cx x 16)
__device__ __forceinline__ v8f wmma_tile(const __bf16* __restrict__ wrow,
                                         const __bf16* __restrict__ xcol,
                                         int Cx, int h) {
  v8f c = {0.f,0.f,0.f,0.f,0.f,0.f,0.f,0.f};
  for (int kb = 0; kb < Cx; kb += 32) {
    if (kb + 32 < Cx) __builtin_prefetch(xcol + kb + 32, 0, 1);  // global_prefetch_b8
    v8bf alo = *(const v8bf*)(wrow + kb + 16 * h);
    v8bf ahi = *(const v8bf*)(wrow + kb + 16 * h + 8);
    v8bf blo = *(const v8bf*)(xcol + kb + 16 * h);
    v8bf bhi = *(const v8bf*)(xcol + kb + 16 * h + 8);
    v16bf a = cat16(alo, ahi);
    v16bf b = cat16(blo, bhi);
    c = __builtin_amdgcn_wmma_f32_16x16x32_bf16(false, a, false, b, (short)0, c,
                                                false, false);
  }
  return c;
}

// ---------------- kNN: one batch's 2048 points cached in LDS (24KB) --------
__global__ __launch_bounds__(256) void knn_kernel(const float* __restrict__ pc,
                                                  int* __restrict__ idx) {
  __shared__ float sp[N_ * 3];
  int b  = blockIdx.x >> 3;          // 8 blocks per batch
  int nb = (blockIdx.x & 7) << 8;
  const float* src = pc + (size_t)b * N_ * 3;
  for (int i = threadIdx.x; i < N_ * 3; i += 256) sp[i] = src[i];
  __syncthreads();
  int n = nb + threadIdx.x;
  float qx = sp[n * 3], qy = sp[n * 3 + 1], qz = sp[n * 3 + 2];
  float val[K1_]; int ind[K1_];
  for (int i = 0; i < K1_; ++i) { val[i] = -3.4e38f; ind[i] = 0; }
  for (int m = 0; m < N_; ++m) {
    float dx = sp[m * 3] - qx, dy = sp[m * 3 + 1] - qy, dz = sp[m * 3 + 2] - qz;
    float d = -(dx * dx + dy * dy + dz * dz);   // neg squared distance
    if (d > val[K1_ - 1]) {
      int j = K1_ - 1;
      while (j > 0 && val[j - 1] < d) { val[j] = val[j - 1]; ind[j] = ind[j - 1]; --j; }
      val[j] = d; ind[j] = m;        // stable: ties keep earlier index first
    }
  }
  int* op = idx + ((size_t)b * N_ + n) * K1_;
  for (int i = 0; i < K1_; ++i) op[i] = ind[i];
}

// ---------------- gather neighbors -> (P,16,32) & (P,8,32) bf16, ch 3..31=0 -
__global__ __launch_bounds__(256) void gather_kernel(const float* __restrict__ pc,
                                                     const int* __restrict__ idx,
                                                     __bf16* __restrict__ X1,
                                                     __bf16* __restrict__ X2) {
  int id = blockIdx.x * 256 + threadIdx.x;           // over P_*K1_
  if (id >= P_ * K1_) return;
  int p = id >> 4, k = id & 15;
  int b = p >> 11;
  int i = idx[id];
  const float* q = pc + ((size_t)b * N_ + i) * 3;
  __bf16* o1 = X1 + (size_t)id * 32;
  #pragma unroll
  for (int c = 0; c < 32; ++c) o1[c] = (__bf16)0.f;
  o1[0] = (__bf16)q[0]; o1[1] = (__bf16)q[1]; o1[2] = (__bf16)q[2];
  if (k < K2_) {
    __bf16* o2 = X2 + ((size_t)p * K2_ + k) * 32;
    #pragma unroll
    for (int c = 0; c < 32; ++c) o2[c] = (__bf16)0.f;
    o2[0] = (__bf16)q[0]; o2[1] = (__bf16)q[1]; o2[2] = (__bf16)q[2];
  }
}

// ---------------- weight f32 -> bf16 (K-padded row-major Cout x Cinp) ------
__global__ void convw_kernel(const float* __restrict__ w, __bf16* __restrict__ o,
                             int Cout, int Cin, int Cinp) {
  int i = blockIdx.x * blockDim.x + threadIdx.x;
  if (i >= Cout * Cinp) return;
  int r = i / Cinp, c = i % Cinp;
  o[i] = (__bf16)(c < Cin ? w[r * Cin + c] : 0.f);
}

__global__ void zero_kernel(float* p, int n) {
  int i = blockIdx.x * blockDim.x + threadIdx.x;
  if (i < n) p[i] = 0.f;
}

// ---------------- pass 1: GEMM tiles -> per-channel sum / sumsq ------------
__global__ __launch_bounds__(256) void gemm_stats_kernel(
    const __bf16* __restrict__ W, const __bf16* __restrict__ X,
    int L, int Cx, int Cout, float* __restrict__ sum, float* __restrict__ sumsq) {
  const int lane = threadIdx.x & 31;
  const int wid  = blockIdx.x * 8 + (threadIdx.x >> 5);
  const int tiles_m = Cout >> 4;
  const int tm   = wid % tiles_m;
  const int row0 = wid / tiles_m;
  const int r = lane & 15, h = lane >> 4;
  const __bf16* wrow = W + (size_t)(tm * 16 + r) * Cx;
  float s[8], sq[8];
  #pragma unroll
  for (int i = 0; i < 8; ++i) { s[i] = 0.f; sq[i] = 0.f; }
  for (int t = 0; t < PT_; ++t) {
    int pbase = (row0 * PT_ + t) * 16;        // wave-uniform bound check
    if (pbase >= L) break;
    const __bf16* xcol = X + (size_t)(pbase + r) * Cx;
    v8f c = wmma_tile(wrow, xcol, Cx, h);
    #pragma unroll
    for (int i = 0; i < 8; ++i) { float v = c[i]; s[i] += v; sq[i] += v * v; }
  }
  #pragma unroll
  for (int m = 1; m < 16; m <<= 1) {
    #pragma unroll
    for (int i = 0; i < 8; ++i) {
      s[i]  += __shfl_xor(s[i],  m, 32);
      sq[i] += __shfl_xor(sq[i], m, 32);
    }
  }
  if (r == 0) {
    int ch0 = tm * 16 + 8 * h;
    #pragma unroll
    for (int i = 0; i < 8; ++i) {
      atomicAdd(&sum[ch0 + i],   s[i]);
      atomicAdd(&sumsq[ch0 + i], sq[i]);
    }
  }
}

// ---------------- finalize BN: scale=g*rsqrt(var+eps), shift=be-mean*scale -
__global__ void finalize_kernel(const float* __restrict__ sum,
                                const float* __restrict__ sumsq,
                                const float* __restrict__ g,
                                const float* __restrict__ be,
                                int C, float invL, float* __restrict__ ss) {
  int c = blockIdx.x * blockDim.x + threadIdx.x;
  if (c >= C) return;
  float mean = sum[c] * invL;
  float var  = sumsq[c] * invL - mean * mean;
  float rs   = rsqrtf(var + EPS_);
  float sc   = g[c] * rs;
  ss[c]     = sc;
  ss[C + c] = be[c] - mean * sc;
}

// ---------------- pass 2: recompute GEMM, fuse scale/shift + ReLU, store ---
__global__ __launch_bounds__(256) void gemm_norm_kernel(
    const __bf16* __restrict__ W, const __bf16* __restrict__ X,
    int L, int Cx, int Cout, const float* __restrict__ ss,
    __bf16* __restrict__ Y) {
  const int lane = threadIdx.x & 31;
  const int wid  = blockIdx.x * 8 + (threadIdx.x >> 5);
  const int tiles_m = Cout >> 4;
  const int tm   = wid % tiles_m;
  const int row0 = wid / tiles_m;
  const int r = lane & 15, h = lane >> 4;
  const int ch0 = tm * 16 + 8 * h;
  const __bf16* wrow = W + (size_t)(tm * 16 + r) * Cx;
  float sc[8], sh[8];
  #pragma unroll
  for (int i = 0; i < 8; ++i) { sc[i] = ss[ch0 + i]; sh[i] = ss[Cout + ch0 + i]; }
  for (int t = 0; t < PT_; ++t) {
    int pbase = (row0 * PT_ + t) * 16;
    if (pbase >= L) break;
    int p = pbase + r;
    const __bf16* xcol = X + (size_t)p * Cx;
    v8f c = wmma_tile(wrow, xcol, Cx, h);
    v8bf o;
    #pragma unroll
    for (int i = 0; i < 8; ++i) {
      float v = fmaxf(c[i] * sc[i] + sh[i], 0.f);
      o[i] = (__bf16)v;
    }
    *(v8bf*)(Y + (size_t)p * Cout + ch0) = o;   // 8 contiguous channels
  }
}

// ---------------- stage-5 variant: f32 output, layout (B, 512, N) ----------
__global__ __launch_bounds__(256) void gemm_norm_out_kernel(
    const __bf16* __restrict__ W, const __bf16* __restrict__ X,
    int L, int Cx, int Cout, const float* __restrict__ ss,
    float* __restrict__ out) {
  const int lane = threadIdx.x & 31;
  const int wid  = blockIdx.x * 8 + (threadIdx.x >> 5);
  const int tiles_m = Cout >> 4;
  const int tm   = wid % tiles_m;
  const int row0 = wid / tiles_m;
  const int r = lane & 15, h = lane >> 4;
  const int ch0 = tm * 16 + 8 * h;
  const __bf16* wrow = W + (size_t)(tm * 16 + r) * Cx;
  float sc[8], sh[8];
  #pragma unroll
  for (int i = 0; i < 8; ++i) { sc[i] = ss[ch0 + i]; sh[i] = ss[Cout + ch0 + i]; }
  for (int t = 0; t < PT_; ++t) {
    int pbase = (row0 * PT_ + t) * 16;
    if (pbase >= L) break;
    int p = pbase + r;                 // p = b*N + n
    const __bf16* xcol = X + (size_t)p * Cx;
    v8f c = wmma_tile(wrow, xcol, Cx, h);
    int b = p >> 11, n = p & (N_ - 1);
    float* op = out + ((size_t)b * Cout + ch0) * N_ + n;
    #pragma unroll
    for (int i = 0; i < 8; ++i)
      op[(size_t)i * N_] = fmaxf(c[i] * sc[i] + sh[i], 0.f);
  }
}

// ---------------- max over K (channel-last), f32 out -----------------------
__global__ void max_kernel(const __bf16* __restrict__ X, int K, int C,
                           float* __restrict__ m, int n) {
  int id = blockIdx.x * blockDim.x + threadIdx.x;
  if (id >= n) return;
  int p = id / C, c = id % C;
  const __bf16* xp = X + (size_t)p * K * C + c;
  float mm = (float)xp[0];
  for (int k = 1; k < K; ++k) mm = fmaxf(mm, (float)xp[(size_t)k * C]);
  m[id] = mm;
}

__global__ void addu_kernel(const float* __restrict__ a, const float* __restrict__ b,
                            __bf16* __restrict__ u, int n) {
  int id = blockIdx.x * blockDim.x + threadIdx.x;
  if (id < n) u[id] = (__bf16)(a[id] + b[id]);
}

// ---------------- fcn second conv (1 x r) + single-channel stats -----------
__global__ __launch_bounds__(256) void dot_stats_kernel(
    const float* __restrict__ w, const __bf16* __restrict__ hbuf, int r,
    float* __restrict__ raw, float* __restrict__ s2) {
  int p = blockIdx.x * 256 + threadIdx.x;
  float acc = 0.f;
  const __bf16* hp = hbuf + (size_t)p * r;
  for (int j = 0; j < r; ++j) acc += w[j] * (float)hp[j];
  raw[p] = acc;
  float a1 = acc, a2 = acc * acc;
  for (int m = 1; m < 32; m <<= 1) { a1 += __shfl_xor(a1, m, 32); a2 += __shfl_xor(a2, m, 32); }
  __shared__ float r1[8], r2[8];
  if ((threadIdx.x & 31) == 0) { r1[threadIdx.x >> 5] = a1; r2[threadIdx.x >> 5] = a2; }
  __syncthreads();
  if (threadIdx.x == 0) {
    float t1 = 0.f, t2 = 0.f;
    for (int i = 0; i < 8; ++i) { t1 += r1[i]; t2 += r2[i]; }
    atomicAdd(&s2[0], t1); atomicAdd(&s2[1], t2);
  }
}

// ---------------- attention fuse: softmax(2) == sigmoid(a-b) ---------------
__global__ void fuse_kernel(const float* __restrict__ araw, const float* __restrict__ ssa,
                            const float* __restrict__ braw, const float* __restrict__ ssb,
                            const float* __restrict__ m1, const float* __restrict__ m2,
                            int C, __bf16* __restrict__ cat, int coff, int n) {
  int id = blockIdx.x * blockDim.x + threadIdx.x;
  if (id >= n) return;
  int p = id / C, c = id % C;
  float an = araw[p] * ssa[0] + ssa[1];
  float bn = braw[p] * ssb[0] + ssb[1];
  float w  = 1.f / (1.f + expf(bn - an));
  cat[(size_t)p * 512 + coff + c] = (__bf16)(w * m1[id] + (1.f - w) * m2[id]);
}

// ===========================================================================
extern "C" void kernel_launch(void* const* d_in, const int* in_sizes, int n_in,
                              void* d_out, int out_size, void* d_ws, size_t ws_size,
                              hipStream_t stream) {
  (void)in_sizes; (void)n_in; (void)out_size; (void)ws_size;
  const float* pc = (const float*)d_in[0];

  // input indexing (setup_inputs dict order):
  // 0 pointcloud | 1..4 conv2d w | 5..12 bn2d (g,b)x4
  // 13+16s+8br+{0..7}: pa[s] branch br = (w1,b1,g1,be1,w2,b2,g2,be2)
  // 77 w5 | 78 b5 | 79 g5 | 80 be5     (biases cancel inside train-mode BN)

  char* base = (char*)d_ws;
  size_t off = 0;
  auto alloc = [&](size_t bytes) -> char* {
    char* p = base + off;
    off = (off + bytes + 255) & ~(size_t)255;
    return p;
  };

  int*    idxb = (int*)   alloc((size_t)P_ * K1_ * 4);
  __bf16* X1a  = (__bf16*)alloc((size_t)L1_ * 256 * 2);
  __bf16* X1b  = (__bf16*)alloc((size_t)L1_ * 128 * 2);
  __bf16* X2a  = (__bf16*)alloc((size_t)L2_ * 256 * 2);
  __bf16* X2b  = (__bf16*)alloc((size_t)L2_ * 128 * 2);
  float*  m1   = (float*) alloc((size_t)P_ * 256 * 4);
  float*  m2   = (float*) alloc((size_t)P_ * 256 * 4);
  __bf16* ub   = (__bf16*)alloc((size_t)P_ * 256 * 2);
  __bf16* hb   = (__bf16*)alloc((size_t)P_ * 64 * 2);
  float*  araw = (float*) alloc((size_t)P_ * 4);
  float*  braw = (float*) alloc((size_t)P_ * 4);
  __bf16* catb = (__bf16*)alloc((size_t)P_ * 512 * 2);
  float*  stats= (float*) alloc(1024 * 4);
  float*  ssn  = (float*) alloc(1024 * 4);
  float*  sc16 = (float*) alloc(64);
  float *s2a = sc16, *s2b = sc16 + 2, *ssa2 = sc16 + 4, *ssb2 = sc16 + 6;

  const int CinA[4]  = {3, 64, 64, 128};
  const int CinP[4]  = {32, 64, 64, 128};
  const int CoutS[4] = {64, 64, 128, 256};
  const int RR[4]    = {16, 16, 32, 64};
  const int COFF[4]  = {0, 64, 128, 256};

  __bf16* wst[4];
  __bf16* wf[4][2];
  for (int s = 0; s < 4; ++s) wst[s] = (__bf16*)alloc((size_t)CoutS[s] * CinP[s] * 2);
  for (int s = 0; s < 4; ++s)
    for (int br = 0; br < 2; ++br) wf[s][br] = (__bf16*)alloc((size_t)RR[s] * CoutS[s] * 2);
  __bf16* w5b = (__bf16*)alloc((size_t)512 * 512 * 2);

  // ---- weight conversion ----
  for (int s = 0; s < 4; ++s) {
    int n = CoutS[s] * CinP[s];
    convw_kernel<<<(n + 255) / 256, 256, 0, stream>>>((const float*)d_in[1 + s],
                                                      wst[s], CoutS[s], CinA[s], CinP[s]);
  }
  for (int s = 0; s < 4; ++s)
    for (int br = 0; br < 2; ++br) {
      int n = RR[s] * CoutS[s];
      convw_kernel<<<(n + 255) / 256, 256, 0, stream>>>(
          (const float*)d_in[13 + 16 * s + 8 * br], wf[s][br], RR[s], CoutS[s], CoutS[s]);
    }
  convw_kernel<<<(512 * 512 + 255) / 256, 256, 0, stream>>>((const float*)d_in[77],
                                                            w5b, 512, 512, 512);

  // ---- kNN + gather ----
  knn_kernel<<<B_ * (N_ / 256), 256, 0, stream>>>(pc, idxb);
  gather_kernel<<<(P_ * K1_ + 255) / 256, 256, 0, stream>>>(pc, idxb, X1a, X2a);

  auto g_stats = [&](const __bf16* W, const __bf16* X, int L, int Cx, int Co) {
    zero_kernel<<<(2 * Co + 255) / 256, 256, 0, stream>>>(stats, 2 * Co);
    long waves = (long)(Co / 16) * ((L / 16 + PT_ - 1) / PT_);
    gemm_stats_kernel<<<(int)((waves + 7) / 8), 256, 0, stream>>>(W, X, L, Cx, Co,
                                                                  stats, stats + Co);
  };
  auto g_fin = [&](const float* g, const float* be, int Co, float invL) {
    finalize_kernel<<<(Co + 255) / 256, 256, 0, stream>>>(stats, stats + Co, g, be,
                                                          Co, invL, ssn);
  };
  auto g_norm = [&](const __bf16* W, const __bf16* X, int L, int Cx, int Co, __bf16* Y) {
    long waves = (long)(Co / 16) * ((L / 16 + PT_ - 1) / PT_);
    gemm_norm_kernel<<<(int)((waves + 7) / 8), 256, 0, stream>>>(W, X, L, Cx, Co, ssn, Y);
  };

  __bf16 *x1i = X1a, *x1o = X1b, *x2i = X2a, *x2o = X2b;
  for (int s = 0; s < 4; ++s) {
    int Ci = CinP[s], Co = CoutS[s], R = RR[s];
    const float* g  = (const float*)d_in[5 + 2 * s];
    const float* be = (const float*)d_in[6 + 2 * s];

    // lf1 stream (K=16)
    g_stats(wst[s], x1i, L1_, Ci, Co);
    g_fin(g, be, Co, 1.0f / (float)L1_);
    g_norm(wst[s], x1i, L1_, Ci, Co, x1o);
    // lf2 stream (K=8)
    g_stats(wst[s], x2i, L2_, Ci, Co);
    g_fin(g, be, Co, 1.0f / (float)L2_);
    g_norm(wst[s], x2i, L2_, Ci, Co, x2o);

    int nPC = P_ * Co;
    max_kernel<<<(nPC + 255) / 256, 256, 0, stream>>>(x1o, K1_, Co, m1, nPC);
    max_kernel<<<(nPC + 255) / 256, 256, 0, stream>>>(x2o, K2_, Co, m2, nPC);
    addu_kernel<<<(nPC + 255) / 256, 256, 0, stream>>>(m1, m2, ub, nPC);

    for (int br = 0; br < 2; ++br) {
      int pb = 13 + 16 * s + 8 * br;
      const float* g1  = (const float*)d_in[pb + 2];
      const float* be1 = (const float*)d_in[pb + 3];
      const float* w2  = (const float*)d_in[pb + 4];
      const float* g2  = (const float*)d_in[pb + 6];
      const float* be2 = (const float*)d_in[pb + 7];
      g_stats(wf[s][br], ub, P_, Co, R);
      g_fin(g1, be1, R, 1.0f / (float)P_);
      g_norm(wf[s][br], ub, P_, Co, R, hb);
      float* raw = br ? braw : araw;
      float* s2  = br ? s2b : s2a;
      float* ssx = br ? ssb2 : ssa2;
      zero_kernel<<<1, 32, 0, stream>>>(s2, 2);
      dot_stats_kernel<<<P_ / 256, 256, 0, stream>>>(w2, hb, R, raw, s2);
      finalize_kernel<<<1, 32, 0, stream>>>(s2, s2 + 1, g2, be2, 1, 1.0f / (float)P_, ssx);
    }
    int nPC2 = P_ * Co;
    fuse_kernel<<<(nPC2 + 255) / 256, 256, 0, stream>>>(araw, ssa2, braw, ssb2,
                                                        m1, m2, Co, catb, COFF[s], nPC2);
    { __bf16* t = x1i; x1i = x1o; x1o = t; t = x2i; x2i = x2o; x2o = t; }
  }

  // ---- stage 5: 512x512 GEMM -> BN -> ReLU -> (B,512,N) f32 ----
  g_stats(w5b, catb, P_, 512, 512);
  g_fin((const float*)d_in[79], (const float*)d_in[80], 512, 1.0f / (float)P_);
  {
    long waves = (long)(512 / 16) * ((P_ / 16 + PT_ - 1) / PT_);
    gemm_norm_out_kernel<<<(int)((waves + 7) / 8), 256, 0, stream>>>(
        w5b, catb, P_, 512, 512, ssn, (float*)d_out);
  }
}
